// ReformerPPBlock_10926396801631
// MI455X (gfx1250) — compile-verified, matrix operand order checked by reading, not run on hardware
//
// Reformer++ block for MI455X (gfx1250, wave32, WMMA).
// All matmuls (5 projection GEMMs + chunked attention QK^T and P*V) run on
// v_wmma_f32_16x16x32_bf16 with f32 accumulation. GEMM register-blocks 4
// N-tiles per wave (one A fragment feeds 4 WMMAs). The LSH attention gathers
// Q/K rows into LDS with GLOBAL_LOAD_ASYNC_TO_LDS_B128 (ASYNCcnt-tracked).
#include <hip/hip_runtime.h>
#include <hip/hip_bf16.h>
#include <math.h>

typedef __bf16 bf16_t;
typedef __attribute__((ext_vector_type(16))) __bf16 v16bf;
typedef __attribute__((ext_vector_type(8)))  __bf16 v8bf;
typedef __attribute__((ext_vector_type(8)))  float  v8f;

#define DEV __device__ __forceinline__

constexpr int Bc = 2, Sc = 4096, Dc = 1024, Hc = 16, DHc = 64;
constexpr int NHc = 4, Cc = 64, NBH = 32, RADc = 4;
constexpr float EPSc = 1e-6f, SCALEc = 0.125f; // DH^-0.5

DEV bf16_t f2bf(float f) {
  unsigned u = __float_as_uint(f);
  unsigned r = (u + 0x7FFFu + ((u >> 16) & 1u)) >> 16;
  unsigned short s = (unsigned short)r;
  return __builtin_bit_cast(bf16_t, s);
}
DEV float sigf(float x) { return 1.f / (1.f + __expf(-x)); }

// Load one 16-bit WMMA operand fragment from a row-major [rows][ld] bf16 array.
// Lane mapping (ISA 7.12.2): lane L holds row L%16, K-half L/16; K elements are
// two contiguous groups of 8 at kbase+half*8 and kbase+16+half*8.
DEV v16bf load_frag(const bf16_t* base, int ld, int row, int kbase, int khalf) {
  const v8bf* p0 = (const v8bf*)(base + (size_t)row * ld + kbase + khalf * 8);
  const v8bf* p1 = (const v8bf*)(base + (size_t)row * ld + kbase + 16 + khalf * 8);
  v8bf lo = *p0, hi = *p1;
  v16bf a;
#pragma unroll
  for (int i = 0; i < 8; ++i) { a[i] = lo[i]; a[i + 8] = hi[i]; }
  return a;
}

DEV v8f wmma_bf16(v16bf a, v16bf b, v8f c) {
  return __builtin_amdgcn_wmma_f32_16x16x32_bf16(false, a, false, b, (short)0, c,
                                                 false, false);
}

// Async gather of one 16-byte chunk from global into LDS (ASYNCcnt-tracked).
DEV void async_lds_b128(const bf16_t* gptr, void* lds_dst) {
  unsigned lo = (unsigned)(uintptr_t)lds_dst;  // LDS aperture: addr[31:0] = offset
  asm volatile("global_load_async_to_lds_b128 %0, %1, off"
               :: "v"(lo), "v"((unsigned long long)(uintptr_t)gptr)
               : "memory");
}

// ---------------- conversion kernels ----------------
__global__ void k_cvt(const float* __restrict__ in, bf16_t* __restrict__ out, size_t n) {
  size_t i = (size_t)blockIdx.x * blockDim.x + threadIdx.x;
  if (i < n) out[i] = f2bf(in[i]);
}
// W [K][N] f32 -> Wt [N][K] bf16 (so B fragments load contiguously along K)
__global__ void k_cvt_t(const float* __restrict__ W, bf16_t* __restrict__ Wt, int K, int N) {
  size_t i = (size_t)blockIdx.x * blockDim.x + threadIdx.x;
  if (i >= (size_t)K * N) return;
  int n = (int)(i / K), k = (int)(i % K);
  Wt[i] = f2bf(W[(size_t)k * N + n]);
}

// ---------------- generic bf16 WMMA GEMM ----------------
// C[M][N] = A[M][K] * Wt[N][K]^T (+bias) (relu?) ; optional f32 and bf16 outputs.
// Each wave computes a 16x64 strip: one A fragment feeds 4 WMMAs (register
// blocking along N cuts L2 traffic ~3x vs 16x16-per-wave).
__global__ void k_gemm(const bf16_t* __restrict__ A, const bf16_t* __restrict__ Bt,
                       const float* __restrict__ bias, float* __restrict__ Cf,
                       bf16_t* __restrict__ Cbf, int M, int N, int K, int relu) {
  int wave = threadIdx.x >> 5, lane = threadIdx.x & 31;
  int l16 = lane & 15, half = lane >> 4;
  int tilesN = N >> 6;  // groups of 64 columns
  int flat = blockIdx.x * 4 + wave;
  int tm = (flat / tilesN) << 4, tn = (flat % tilesN) << 6;
  if (tm >= M) return;
  int mrow = tm + l16;
  v8f zero = {0.f, 0.f, 0.f, 0.f, 0.f, 0.f, 0.f, 0.f};
  v8f acc[4];
#pragma unroll
  for (int t = 0; t < 4; ++t) acc[t] = zero;
  for (int kb = 0; kb < K; kb += 32) {
    if (kb + 64 < K) {
      __builtin_prefetch(A + (size_t)mrow * K + kb + 64, 0, 1);
#pragma unroll
      for (int t = 0; t < 4; ++t)
        __builtin_prefetch(Bt + (size_t)(tn + t * 16 + l16) * K + kb + 64, 0, 1);
    }
    v16bf a = load_frag(A, K, mrow, kb, half);
#pragma unroll
    for (int t = 0; t < 4; ++t) {
      v16bf b = load_frag(Bt, K, tn + t * 16 + l16, kb, half);
      acc[t] = wmma_bf16(a, b, acc[t]);
    }
  }
#pragma unroll
  for (int t = 0; t < 4; ++t) {
    int ncol = tn + t * 16 + l16;
    float bv = bias ? bias[ncol] : 0.f;
#pragma unroll
    for (int r = 0; r < 8; ++r) {
      int m = tm + r + 8 * half;
      float val = acc[t][r] + bv;
      if (relu) val = fmaxf(val, 0.f);
      size_t o = (size_t)m * N + ncol;
      if (Cf)  Cf[o]  = val;
      if (Cbf) Cbf[o] = f2bf(val);
    }
  }
}

// ---------------- k-normalization ----------------
__global__ void k_knorm(const float* __restrict__ qk, float* __restrict__ kf,
                        bf16_t* __restrict__ kbf) {
  int t = blockIdx.x * blockDim.x + threadIdx.x;
  if (t >= Bc * Hc * Sc) return;
  int s = t % Sc, h = (t / Sc) % Hc, b = t / (Sc * Hc);
  size_t off = ((size_t)b * Sc + s) * Dc + h * DHc;
  float ss = 0.f;
  for (int d = 0; d < DHc; ++d) { float q = qk[off + d]; ss += q * q; }
  float inv = 1.f / (sqrtf(ss) + EPSc);
  for (int d = 0; d < DHc; ++d) {
    float kv = qk[off + d] * inv;
    kf[off + d] = kv;
    kbf[off + d] = f2bf(kv);
  }
}

// ---------------- LSH bucketing ----------------
__global__ void k_buckets(const float* __restrict__ qk, const float* __restrict__ rot,
                          int* __restrict__ buckets) {
  int t = blockIdx.x * blockDim.x + threadIdx.x;
  if (t >= Bc * Hc * NHc * Sc) return;
  int s = t % Sc, r = (t / Sc) % NHc, h = (t / (Sc * NHc)) % Hc, b = t / (Sc * NHc * Hc);
  const float* q = qk + ((size_t)b * Sc + s) * Dc + h * DHc;
  float best = -1e30f; int bi = 0;
  for (int m = 0; m < NBH; ++m) {
    float acc = 0.f;
    for (int d = 0; d < DHc; ++d)
      acc += q[d] * rot[(((size_t)h * NHc + r) * DHc + d) * NBH + m];
    if (acc > best)  { best = acc;  bi = m; }
    if (-acc > best) { best = -acc; bi = m + NBH; }
  }
  buckets[t] = bi;
}

// ---------------- stable counting sort per (b,h,hash) segment ----------------
__global__ void k_sort(const int* __restrict__ buckets, int* __restrict__ perm,
                       int* __restrict__ inv) {
  __shared__ int hist[64];
  __shared__ int base[64];
  int seg = blockIdx.x, tid = threadIdx.x;
  const int* bk = buckets + (size_t)seg * Sc;
  if (tid < 64) hist[tid] = 0;
  __syncthreads();
  for (int i = tid; i < Sc; i += blockDim.x) atomicAdd(&hist[bk[i]], 1);
  __syncthreads();
  if (tid == 0) {
    int acc = 0;
    for (int q = 0; q < 64; ++q) { base[q] = acc; acc += hist[q]; }
  }
  __syncthreads();
  if (tid < 64) {
    int cur = base[tid];
    for (int i = 0; i < Sc; ++i)
      if (bk[i] == tid) {
        perm[(size_t)seg * Sc + cur] = i;
        inv[(size_t)seg * Sc + i] = cur;
        cur++;
      }
  }
}

// ---------------- chunked LSH attention (WMMA + async LDS gather) ----------------
// block = 128 threads (4 waves), one block per (b,h,hash,chunk).
__global__ void k_lsh(const bf16_t* __restrict__ qk_bf, const bf16_t* __restrict__ k_bf,
                      const bf16_t* __restrict__ v_bf, const int* __restrict__ perm,
                      float* __restrict__ o_g, float* __restrict__ lse_g) {
  __shared__ __align__(16) bf16_t sq[64 * 64];     // queries  [w][d]
  __shared__ __align__(16) bf16_t k2s[128 * 64];   // keys     [j][d]  (chunk + prev chunk)
  __shared__ __align__(16) bf16_t v2t[64 * 128];   // values^T [d][j]
  __shared__ __align__(16) bf16_t pm[64 * 128];    // probs    [w][j]
  __shared__ int posA[64], posB[64];

  int bid = blockIdx.x;
  int c = bid % Cc, r = (bid / Cc) % NHc, h = (bid / (Cc * NHc)) % Hc,
      b = bid / (Cc * NHc * Hc);
  int cp = (c + Cc - 1) % Cc;
  int seg = (b * Hc + h) * NHc + r;
  const int* pr = perm + (size_t)seg * Sc;
  int tid = threadIdx.x;

  if (tid < 64)        posA[tid] = pr[c * 64 + tid];
  else if (tid < 128)  posB[tid - 64] = pr[cp * 64 + (tid - 64)];
  __syncthreads();

  // async gather: Q rows (64 x 128B) and K rows (128 x 128B) in 16B chunks;
  // data goes global->LDS without touching VGPRs (ASYNCcnt-tracked).
  for (int idx = tid; idx < 64 * 8; idx += 128) {
    int row = idx >> 3, ch = idx & 7;
    const bf16_t* g = qk_bf + ((size_t)b * Sc + posA[row]) * Dc + h * DHc + ch * 8;
    async_lds_b128(g, &sq[row * 64 + ch * 8]);
  }
  for (int idx = tid; idx < 128 * 8; idx += 128) {
    int row = idx >> 3, ch = idx & 7;
    int p = row < 64 ? posA[row] : posB[row - 64];
    const bf16_t* g = k_bf + ((size_t)b * Sc + p) * Dc + h * DHc + ch * 8;
    async_lds_b128(g, &k2s[row * 64 + ch * 8]);
  }
  // transposed V gather is an inherent 2-byte scatter: scalar path
  for (int idx = tid; idx < 64 * 128; idx += 128) {
    int d = idx >> 7, j = idx & 127;
    int p = j < 64 ? posA[j] : posB[j - 64];
    v2t[(size_t)d * 128 + j] = v_bf[((size_t)b * Sc + p) * Dc + h * DHc + d];
  }
  asm volatile("s_wait_asynccnt 0x0" ::: "memory");
  __syncthreads();

  int wv = tid >> 5, lane = tid & 31;
  int l16 = lane & 15, half = lane >> 4;
  int mrow = wv * 16 + l16;

  // dots[64][128] = sq * k2^T, accumulated in registers (8 column tiles)
  v8f dacc[8];
  v8f zero = {0.f, 0.f, 0.f, 0.f, 0.f, 0.f, 0.f, 0.f};
#pragma unroll
  for (int t = 0; t < 8; ++t) dacc[t] = zero;
  for (int kb = 0; kb < 64; kb += 32) {
    v16bf a = load_frag(sq, 64, mrow, kb, half);
#pragma unroll
    for (int t = 0; t < 8; ++t) {
      v16bf bb = load_frag(k2s, 64, t * 16 + l16, kb, half);
      dacc[t] = wmma_bf16(a, bb, dacc[t]);
    }
  }

  // logsumexp per row: rows live across a 16-lane half-group in the C layout.
#pragma unroll
  for (int rr = 0; rr < 8; ++rr) {
    float mx = -1e30f;
#pragma unroll
    for (int t = 0; t < 8; ++t) mx = fmaxf(mx, dacc[t][rr] * SCALEc);
    for (int off = 1; off < 16; off <<= 1) mx = fmaxf(mx, __shfl_xor(mx, off));
    float sum = 0.f;
#pragma unroll
    for (int t = 0; t < 8; ++t) sum += __expf(dacc[t][rr] * SCALEc - mx);
    for (int off = 1; off < 16; off <<= 1) sum += __shfl_xor(sum, off);
    float lse = mx + __logf(sum);
    int row = wv * 16 + rr + 8 * half;
#pragma unroll
    for (int t = 0; t < 8; ++t) {
      float p = __expf(dacc[t][rr] * SCALEc - lse);
      pm[(size_t)row * 128 + t * 16 + l16] = f2bf(p);
    }
    if (l16 == 0) lse_g[(size_t)seg * Sc + c * 64 + row] = lse;
  }
  __syncthreads();

  // o[64][64] = p * v2  (K = 128 keys)
  v8f oacc[4];
#pragma unroll
  for (int t = 0; t < 4; ++t) oacc[t] = zero;
  for (int kb = 0; kb < 128; kb += 32) {
    v16bf a = load_frag(pm, 128, mrow, kb, half);
#pragma unroll
    for (int t = 0; t < 4; ++t) {
      v16bf bb = load_frag(v2t, 128, t * 16 + l16, kb, half);
      oacc[t] = wmma_bf16(a, bb, oacc[t]);
    }
  }
#pragma unroll
  for (int t = 0; t < 4; ++t)
#pragma unroll
    for (int rr = 0; rr < 8; ++rr) {
      int row = wv * 16 + rr + 8 * half;
      o_g[((size_t)seg * Sc + c * 64 + row) * DHc + t * 16 + l16] = oacc[t][rr];
    }
}

// ---------------- unsort + hash-mix + local attention + gate ----------------
__global__ void k_zero(float* p) { if (threadIdx.x == 0 && blockIdx.x == 0) *p = 0.f; }

__global__ void k_combine(const float* __restrict__ qk, const float* __restrict__ kf,
                          const float* __restrict__ vf, const float* __restrict__ o_g,
                          const float* __restrict__ lse_g, const int* __restrict__ inv,
                          const float* __restrict__ w_gate, const float* __restrict__ b_gate,
                          bf16_t* __restrict__ mix_bf, float* __restrict__ reg_acc) {
  int t = blockIdx.x * blockDim.x + threadIdx.x;
  if (t >= Bc * Hc * Sc) return;
  int s = t % Sc, h = (t / Sc) % Hc, b = t / (Sc * Hc);
  size_t rowq = ((size_t)b * Sc + s) * Dc + h * DHc;
  float q[DHc];
  for (int d = 0; d < DHc; ++d) q[d] = qk[rowq + d];

  // softmax over hashes of unsorted lse
  float l[NHc]; int jidx[NHc];
  for (int r = 0; r < NHc; ++r) {
    int seg = (b * Hc + h) * NHc + r;
    int j = inv[(size_t)seg * Sc + s];
    jidx[r] = j;
    l[r] = lse_g[(size_t)seg * Sc + j];
  }
  float mx = fmaxf(fmaxf(l[0], l[1]), fmaxf(l[2], l[3]));
  float w[NHc], wsum = 0.f;
  for (int r = 0; r < NHc; ++r) { w[r] = __expf(l[r] - mx); wsum += w[r]; }
  for (int r = 0; r < NHc; ++r) w[r] /= wsum;

  // local attention probabilities (9 taps)
  float ds[2 * RADc + 1], dmax = -1e30f;
  for (int wi = 0; wi <= 2 * RADc; ++wi) {
    int idx = s - RADc + wi;
    if (idx < 0 || idx >= Sc) { ds[wi] = -1e30f; continue; }
    const float* kr = kf + ((size_t)b * Sc + idx) * Dc + h * DHc;
    float acc = 0.f;
    for (int d = 0; d < DHc; ++d) acc += q[d] * kr[d];
    ds[wi] = acc * SCALEc;
    dmax = fmaxf(dmax, ds[wi]);
  }
  float p9[2 * RADc + 1], psum = 0.f;
  for (int wi = 0; wi <= 2 * RADc; ++wi) {
    p9[wi] = (ds[wi] < -1e29f) ? 0.f : __expf(ds[wi] - dmax);
    psum += p9[wi];
  }
  for (int wi = 0; wi <= 2 * RADc; ++wi) p9[wi] /= psum;

  float ga = 0.f;
  for (int d = 0; d < DHc; ++d) ga += q[d] * w_gate[h * DHc + d];
  float g = sigf(ga + b_gate[h]);

  for (int d = 0; d < DHc; ++d) {
    float loc = 0.f;
    for (int wi = 0; wi <= 2 * RADc; ++wi)
      if (p9[wi] > 0.f)
        loc += p9[wi] * vf[((size_t)b * Sc + (s - RADc + wi)) * Dc + h * DHc + d];
    float lsh = 0.f;
    for (int r = 0; r < NHc; ++r) {
      int seg = (b * Hc + h) * NHc + r;
      lsh += w[r] * o_g[((size_t)seg * Sc + jidx[r]) * DHc + d];
    }
    mix_bf[rowq + d] = f2bf(g * loc + (1.f - g) * lsh);
  }
  atomicAdd(reg_acc, g * (1.f - g) * (1.f / (Bc * Hc * Sc)));
}

// ---------------- residual + layernorm ----------------
__global__ void k_res_ln(const float* __restrict__ x1, const float* __restrict__ f_out,
                         const float* __restrict__ alpha, const float* __restrict__ ln_g,
                         const float* __restrict__ ln_b, float* __restrict__ y1_out,
                         bf16_t* __restrict__ hn_bf) {
  __shared__ float red[256];
  int row = blockIdx.x, tid = threadIdx.x;
  size_t base = (size_t)row * Dc;
  float v[4], s = 0.f;
#pragma unroll
  for (int j = 0; j < 4; ++j) {
    int d = tid + j * 256;
    float val = x1[base + d] + sigf(alpha[d]) * f_out[base + d];
    y1_out[base + d] = val;
    v[j] = val; s += val;
  }
  red[tid] = s; __syncthreads();
  for (int o = 128; o > 0; o >>= 1) { if (tid < o) red[tid] += red[tid + o]; __syncthreads(); }
  float mean = red[0] / Dc; __syncthreads();
  float s2 = 0.f;
#pragma unroll
  for (int j = 0; j < 4; ++j) { float dm = v[j] - mean; s2 += dm * dm; }
  red[tid] = s2; __syncthreads();
  for (int o = 128; o > 0; o >>= 1) { if (tid < o) red[tid] += red[tid + o]; __syncthreads(); }
  float rs = rsqrtf(red[0] / Dc + 1e-5f);
#pragma unroll
  for (int j = 0; j < 4; ++j) {
    int d = tid + j * 256;
    hn_bf[base + d] = f2bf((v[j] - mean) * rs * ln_g[d] + ln_b[d]);
  }
}

__global__ void k_y2(const float* __restrict__ x2, const float* __restrict__ h2,
                     const float* __restrict__ beta, float* __restrict__ y2) {
  size_t i = (size_t)blockIdx.x * blockDim.x + threadIdx.x;
  if (i >= (size_t)Bc * Sc * Dc) return;
  int d = (int)(i % Dc);
  y2[i] = x2[i] + sigf(beta[d]) * h2[i];
}

// ---------------- host launcher ----------------
extern "C" void kernel_launch(void* const* d_in, const int* in_sizes, int n_in,
                              void* d_out, int out_size, void* d_ws, size_t ws_size,
                              hipStream_t stream) {
  (void)in_sizes; (void)n_in; (void)out_size; (void)ws_size;
  const float* x1     = (const float*)d_in[0];
  const float* x2     = (const float*)d_in[1];
  const float* Wqk    = (const float*)d_in[2];
  const float* Wv     = (const float*)d_in[3];
  const float* Wo     = (const float*)d_in[4];
  const float* rot    = (const float*)d_in[5];
  const float* w_gate = (const float*)d_in[6];
  const float* b_gate = (const float*)d_in[7];
  const float* ln_g   = (const float*)d_in[8];
  const float* ln_b   = (const float*)d_in[9];
  const float* W1     = (const float*)d_in[10];
  const float* b1     = (const float*)d_in[11];
  const float* W2     = (const float*)d_in[12];
  const float* b2     = (const float*)d_in[13];
  const float* alpha  = (const float*)d_in[14];
  const float* beta   = (const float*)d_in[15];

  const size_t BSD = (size_t)Bc * Sc * Dc;           // 8388608
  const int M = Bc * Sc;                              // 8192
  float* out = (float*)d_out;
  float* y1_out = out;
  float* y2_out = out + BSD;
  float* reg_out = out + 2 * BSD;

  char* ws = (char*)d_ws; size_t off = 0;
  auto alloc = [&](size_t bytes) -> void* {
    off = (off + 255) & ~(size_t)255;
    void* p = ws + off; off += bytes; return p;
  };
  bf16_t* x2_bf  = (bf16_t*)alloc(BSD * 2);
  bf16_t* Wqkt   = (bf16_t*)alloc((size_t)Dc * Dc * 2);
  bf16_t* Wvt    = (bf16_t*)alloc((size_t)Dc * Dc * 2);
  bf16_t* Wot    = (bf16_t*)alloc((size_t)Dc * Dc * 2);
  bf16_t* W1t    = (bf16_t*)alloc((size_t)Dc * Dc * 2);
  bf16_t* W2t    = (bf16_t*)alloc((size_t)Dc * Dc * 2);
  float*  qk_f   = (float*)alloc(BSD * 4);
  float*  v_f    = (float*)alloc(BSD * 4);
  bf16_t* qk_bf  = (bf16_t*)alloc(BSD * 2);
  bf16_t* v_bf   = (bf16_t*)alloc(BSD * 2);
  float*  k_f    = (float*)alloc(BSD * 4);
  bf16_t* k_bf   = (bf16_t*)alloc(BSD * 2);
  int*    bkt    = (int*)alloc((size_t)Bc * Hc * NHc * Sc * 4);
  int*    perm   = (int*)alloc((size_t)Bc * Hc * NHc * Sc * 4);
  int*    invp   = (int*)alloc((size_t)Bc * Hc * NHc * Sc * 4);
  float*  o_g    = (float*)alloc((size_t)Bc * Hc * NHc * Sc * DHc * 4);
  float*  lse_g  = (float*)alloc((size_t)Bc * Hc * NHc * Sc * 4);
  bf16_t* mix_bf = (bf16_t*)alloc(BSD * 2);
  float*  f_out  = (float*)alloc(BSD * 4);
  bf16_t* hn_bf  = (bf16_t*)alloc(BSD * 2);
  bf16_t* h1_bf  = (bf16_t*)alloc(BSD * 2);
  float*  h2_f   = (float*)alloc(BSD * 4);

  // conversions
  k_cvt<<<(int)((BSD + 255) / 256), 256, 0, stream>>>(x2, x2_bf, BSD);
  int nW = Dc * Dc;
  k_cvt_t<<<(nW + 255) / 256, 256, 0, stream>>>(Wqk, Wqkt, Dc, Dc);
  k_cvt_t<<<(nW + 255) / 256, 256, 0, stream>>>(Wv,  Wvt,  Dc, Dc);
  k_cvt_t<<<(nW + 255) / 256, 256, 0, stream>>>(Wo,  Wot,  Dc, Dc);
  k_cvt_t<<<(nW + 255) / 256, 256, 0, stream>>>(W1,  W1t,  Dc, Dc);
  k_cvt_t<<<(nW + 255) / 256, 256, 0, stream>>>(W2,  W2t,  Dc, Dc);

  int gemm_blocks = (M / 16) * (Dc / 64) / 4;  // 16x64 strip per wave, 4 waves/block
  // qk = x2 @ Wqk ; v = x2 @ Wv
  k_gemm<<<gemm_blocks, 128, 0, stream>>>(x2_bf, Wqkt, nullptr, qk_f, qk_bf, M, Dc, Dc, 0);
  k_gemm<<<gemm_blocks, 128, 0, stream>>>(x2_bf, Wvt,  nullptr, v_f,  v_bf,  M, Dc, Dc, 0);

  k_knorm<<<(Bc * Hc * Sc + 255) / 256, 256, 0, stream>>>(qk_f, k_f, k_bf);
  k_buckets<<<(Bc * Hc * NHc * Sc + 255) / 256, 256, 0, stream>>>(qk_f, rot, bkt);
  k_sort<<<Bc * Hc * NHc, 256, 0, stream>>>(bkt, perm, invp);
  k_lsh<<<Bc * Hc * NHc * Cc, 128, 0, stream>>>(qk_bf, k_bf, v_bf, perm, o_g, lse_g);

  k_zero<<<1, 1, 0, stream>>>(reg_out);
  k_combine<<<(Bc * Hc * Sc + 255) / 256, 256, 0, stream>>>(
      qk_f, k_f, v_f, o_g, lse_g, invp, w_gate, b_gate, mix_bf, reg_out);

  // f_out = mix @ Wo
  k_gemm<<<gemm_blocks, 128, 0, stream>>>(mix_bf, Wot, nullptr, f_out, nullptr, M, Dc, Dc, 0);
  k_res_ln<<<M, 256, 0, stream>>>(x1, f_out, alpha, ln_g, ln_b, y1_out, hn_bf);
  // h1 = relu(hn @ W1 + b1) ; h2 = h1 @ W2 + b2
  k_gemm<<<gemm_blocks, 128, 0, stream>>>(hn_bf, W1t, b1, nullptr, h1_bf, M, Dc, Dc, 1);
  k_gemm<<<gemm_blocks, 128, 0, stream>>>(h1_bf, W2t, b2, h2_f, nullptr, M, Dc, Dc, 0);
  k_y2<<<(int)((BSD + 255) / 256), 256, 0, stream>>>(x2, h2_f, beta, y2_out);
}
